// TemporalHead_83434034692798
// MI455X (gfx1250) — matrix operand order repaired and checked by reading, not compiled, not run on hardware
//
#include <hip/hip_runtime.h>
#include <hip/hip_bf16.h>

// ---------------------------------------------------------------------------
// Types for CDNA5 WMMA (wave32) and TDM descriptors
// ---------------------------------------------------------------------------
typedef __bf16 bf16_t;
typedef __attribute__((ext_vector_type(16))) __bf16 v16bf;
typedef __attribute__((ext_vector_type(8)))  __bf16 v8bf;
typedef __attribute__((ext_vector_type(8)))  float  v8f;
typedef __attribute__((ext_vector_type(4)))  unsigned int u32x4;
typedef __attribute__((ext_vector_type(4)))  int i32x4;
typedef __attribute__((ext_vector_type(8)))  int i32x8;

static constexpr int BB = 4;
static constexpr int TT = 4096;
static constexpr int CC = 1024;
static constexpr int HH = 64;

// Native f32 -> bf16 convert (gfx1250 lowers this to v_cvt_*bf16* ops).
static __device__ __forceinline__ bf16_t f2bf(float f) { return (bf16_t)f; }

static __device__ __forceinline__ v16bf cat16(v8bf a, v8bf b) {
    v16bf r;
#pragma unroll
    for (int i = 0; i < 8; ++i) { r[i] = a[i]; r[i + 8] = b[i]; }
    return r;
}

// xor-swizzle within 32 lanes (masks 1,2,4,8 stay inside each 16-lane half)
template <int M>
static __device__ __forceinline__ float swz_xor(float v) {
    int i = __builtin_amdgcn_ds_swizzle(__builtin_bit_cast(int, v),
                                        (M << 10) | 0x1F);
    return __builtin_bit_cast(float, i);
}

static __device__ __forceinline__ v8f wmma_bf16(v16bf a, v16bf b, v8f c) {
    return __builtin_amdgcn_wmma_f32_16x16x32_bf16(
        false, a, false, b, (short)0, c, false, false);
}

// ---------------------------------------------------------------------------
// Kernel 1: fused QKV projection, f32 in -> bf16 out (GEMM M=B*T, N=3*64, K=1024)
// 256 threads = 8 waves; wave w owns rows [blk*128 + 16w, +16) and all 192 cols.
// ---------------------------------------------------------------------------
__global__ __launch_bounds__(256) void qkv_proj_kernel(
    const float* __restrict__ x,
    const float* __restrict__ Wq, const float* __restrict__ Wk,
    const float* __restrict__ Wv,
    bf16_t* __restrict__ Qo, bf16_t* __restrict__ Ko, bf16_t* __restrict__ Vo)
{
    constexpr int KSTEP = 32;
    constexpr int WSTR  = 40;                       // padded LDS row stride (bf16)
    __shared__ __align__(16) bf16_t ldsW[192 * WSTR];

    const int tid  = threadIdx.x;
    const int wave = tid >> 5;
    const int lane = tid & 31;
    const int l16  = lane & 15;
    const int hi   = lane >> 4;                     // 0/1: which lane half
    const int mrow0 = blockIdx.x * 128 + wave * 16;

    v8f acc[12];
    const v8f zero8 = {0.f, 0.f, 0.f, 0.f, 0.f, 0.f, 0.f, 0.f};
#pragma unroll
    for (int i = 0; i < 12; ++i) acc[i] = zero8;

    const float* xrow = x + (size_t)(mrow0 + l16) * CC;
    const int koff = hi ? 8 : 0;                    // A-operand K sub-offset
    const int ks   = hi ? 8 : 0;                    // B-operand K sub-offset

    for (int kk = 0; kk < CC; kk += KSTEP) {
        // ---- cooperative W chunk -> LDS (f32 -> bf16), rows 0..191 ----
        if (tid < 192) {
            const int m = tid >> 6;                 // 0=q 1=k 2=v
            const int h = tid & 63;
            const float* wr = ((m == 0) ? Wq : (m == 1) ? Wk : Wv)
                              + (size_t)h * CC + kk;
            bf16_t* dst = &ldsW[tid * WSTR];
#pragma unroll
            for (int j = 0; j < 32; j += 4) {
                float4 f = *(const float4*)(wr + j);
                dst[j + 0] = f2bf(f.x); dst[j + 1] = f2bf(f.y);
                dst[j + 2] = f2bf(f.z); dst[j + 3] = f2bf(f.w);
            }
        }
        __syncthreads();

        // ---- A operand: 16x32 bf16 tile of x, loaded in WMMA lane layout ----
        v16bf a;
        {
            const float* p0 = xrow + kk + koff;          // K = koff..koff+7
            const float* p1 = xrow + kk + 16 + koff;     // K = 16+koff..+7
#pragma unroll
            for (int j = 0; j < 8; j += 4) {
                float4 f0 = *(const float4*)(p0 + j);
                float4 f1 = *(const float4*)(p1 + j);
                a[j + 0] = f2bf(f0.x); a[j + 1] = f2bf(f0.y);
                a[j + 2] = f2bf(f0.z); a[j + 3] = f2bf(f0.w);
                a[8 + j + 0] = f2bf(f1.x); a[8 + j + 1] = f2bf(f1.y);
                a[8 + j + 2] = f2bf(f1.z); a[8 + j + 3] = f2bf(f1.w);
            }
        }

        // ---- 12 WMMAs: q/k/v x 4 column tiles ----
#pragma unroll
        for (int m = 0; m < 3; ++m) {
#pragma unroll
            for (int nt = 0; nt < 4; ++nt) {
                const bf16_t* bp = &ldsW[(m * 64 + nt * 16 + l16) * WSTR + ks];
                v16bf bm = cat16(*(const v8bf*)bp, *(const v8bf*)(bp + 16));
                acc[m * 4 + nt] = wmma_bf16(a, bm, acc[m * 4 + nt]);
            }
        }
        __syncthreads();
    }

    // ---- store bf16 results: lane holds col = nt*16+l16, rows r(+8 if hi) ----
#pragma unroll
    for (int m = 0; m < 3; ++m) {
        bf16_t* outp = (m == 0) ? Qo : (m == 1) ? Ko : Vo;
#pragma unroll
        for (int nt = 0; nt < 4; ++nt) {
            v8f c = acc[m * 4 + nt];
            const int h = nt * 16 + l16;
#pragma unroll
            for (int r = 0; r < 8; ++r) {
                const int row = mrow0 + r + hi * 8;
                outp[(size_t)row * HH + h] = f2bf(c[r]);
            }
        }
    }
}

// ---------------------------------------------------------------------------
// Kernel 2: causal flash attention over bf16 Q/K/V, f32 output [B,T,H].
// 1 block = 128 query rows (8 waves x 16). 32-key tiles streamed through LDS.
// K tile is DMA'd by the Tensor Data Mover (pad feature provides the +8 bf16
// per-row LDS padding); V tile is transposed cooperatively.
// ---------------------------------------------------------------------------
__global__ __launch_bounds__(256) void flash_attn_kernel(
    const bf16_t* __restrict__ Q, const bf16_t* __restrict__ K,
    const bf16_t* __restrict__ V, float* __restrict__ out)
{
    constexpr int KPAD = 72;                         // kt row stride (= 64 + TDM pad)
    constexpr int VPAD = 40;                         // vt row stride (bf16)
    constexpr int PSTR = 40;                         // P scratch row stride
    __shared__ __align__(16) bf16_t kt[32 * KPAD];   // [key][h]   (TDM-filled)
    __shared__ __align__(16) bf16_t vt[64 * VPAD];   // [h][key]   (transposed)
    __shared__ __align__(16) bf16_t pscr[8 * 16 * PSTR];

    const int tid  = threadIdx.x;
    const int wave = tid >> 5;
    const int lane = tid & 31;
    const int l16  = lane & 15;
    const int hi   = lane >> 4;
    const int b    = blockIdx.x >> 5;                // 32 row-tiles per batch
    const int tb   = blockIdx.x & 31;
    const int qrow0 = tb * 128;
    const int qr    = qrow0 + wave * 16;             // this wave's first row
    const size_t base = (size_t)b * TT * HH;

    // ---- Q tile (16x64) into A-operand registers, two K=32 chunks ----
    const int koff = hi ? 8 : 0;
    const bf16_t* qp = Q + base + (size_t)(qr + l16) * HH;
    const v16bf qa0 = cat16(*(const v8bf*)(qp + koff),
                            *(const v8bf*)(qp + 16 + koff));
    const v16bf qa1 = cat16(*(const v8bf*)(qp + 32 + koff),
                            *(const v8bf*)(qp + 48 + koff));

    const v8f zero8 = {0.f, 0.f, 0.f, 0.f, 0.f, 0.f, 0.f, 0.f};
    v8f o[4];
#pragma unroll
    for (int t = 0; t < 4; ++t) o[t] = zero8;
    float mrun[8], lrun[8];
#pragma unroll
    for (int r = 0; r < 8; ++r) { mrun[r] = -3.0e38f; lrun[r] = 0.f; }

    const int key_hi = qrow0 + 128;                  // exclusive causal bound
    const int ckey = tid >> 3;                       // coop-fill key row
    const int chs  = (tid & 7) * 8;                  // coop-fill h segment
    const unsigned kt_lds = (unsigned)(unsigned long long)(uintptr_t)kt;

    for (int s0 = 0; s0 < key_hi; s0 += 32) {
        // ---- K tile via Tensor Data Mover: 32x64 bf16, pad 4 dw / 32 dw ----
        if (wave == 0) {
            const unsigned long long gaddr =
                (unsigned long long)(uintptr_t)(K + base + (size_t)s0 * HH);
            u32x4 g0;
            g0[0] = 1u;                                    // count=1, no gather
            g0[1] = kt_lds;                                // lds_addr
            g0[2] = (unsigned)gaddr;                       // global_addr[31:0]
            g0[3] = (unsigned)((gaddr >> 32) & 0x1FFFFFFu) // global_addr[56:32]
                    | (2u << 30);                          // type = 2 (image)
            i32x8 g1;
            g1[0] = (1 << 16)                              // data_size = 2 bytes
                  | (1 << 20)                              // pad_enable
                  | (4 << 22)                              // pad_interval: 32 dw
                  | (3 << 25);                             // pad_amount: 4 dw
            g1[1] = (64 << 16);                            // tensor_dim0 = 64 (H)
            g1[2] = (32 << 16);                            // tensor_dim1 = 32 rows
            g1[3] = (64 << 16);                            // tile_dim0 = 64
            g1[4] = 32;                                    // tile_dim1 = 32
            g1[5] = 64;                                    // tensor_dim0_stride
            g1[6] = 0; g1[7] = 0;
            const i32x4 z4 = {0, 0, 0, 0};
            const i32x8 z8 = {0, 0, 0, 0, 0, 0, 0, 0};
            __builtin_amdgcn_tensor_load_to_lds(g0, g1, z4, z4, z8, 0);
        }
        // ---- V tile cooperative transpose fill ----
        {
            const bf16_t* vp = V + base + (size_t)(s0 + ckey) * HH + chs;
            v8bf vv = *(const v8bf*)vp;
#pragma unroll
            for (int j = 0; j < 8; ++j)
                vt[(chs + j) * VPAD + ckey] = vv[j];       // transpose in LDS
            if (s0 + 32 < key_hi)                          // prefetch next tile
                __builtin_prefetch(vp + 32 * HH, 0, 1);
        }
        if (wave == 0)
            __builtin_amdgcn_s_wait_tensorcnt(0);
        __syncthreads();

        if (s0 <= qr + 15) {                          // wave-uniform guard
            // ---- S = (Q K^T) * scale + causal mask, two 16x16 tiles ----
            float sc[2][8];
#pragma unroll
            for (int nb = 0; nb < 2; ++nb) {
                const bf16_t* kb = &kt[(nb * 16 + l16) * KPAD + (hi ? 8 : 0)];
                v16bf b0 = cat16(*(const v8bf*)kb,        *(const v8bf*)(kb + 16));
                v16bf b1 = cat16(*(const v8bf*)(kb + 32), *(const v8bf*)(kb + 48));
                v8f s = zero8;
                s = wmma_bf16(qa0, b0, s);                 // h = 0..31
                s = wmma_bf16(qa1, b1, s);                 // h = 32..63
                const int col = s0 + nb * 16 + l16;
#pragma unroll
                for (int r = 0; r < 8; ++r) {
                    const int row = qr + r + hi * 8;
                    float v = s[r] * 0.125f;               // 1/sqrt(64)
                    sc[nb][r] = (col > row) ? -1.0e9f : v;
                }
            }

            // ---- online softmax (row stats live replicated in half-wave) ----
            float alpha[8];
#pragma unroll
            for (int r = 0; r < 8; ++r) {
                float mx = fmaxf(sc[0][r], sc[1][r]);
                mx = fmaxf(mx, swz_xor<1>(mx));
                mx = fmaxf(mx, swz_xor<2>(mx));
                mx = fmaxf(mx, swz_xor<4>(mx));
                mx = fmaxf(mx, swz_xor<8>(mx));
                const float mnew = fmaxf(mrun[r], mx);
                const float a = __expf(mrun[r] - mnew);
                const float p0 = __expf(sc[0][r] - mnew);
                const float p1 = __expf(sc[1][r] - mnew);
                float rs = p0 + p1;
                rs += swz_xor<1>(rs);
                rs += swz_xor<2>(rs);
                rs += swz_xor<4>(rs);
                rs += swz_xor<8>(rs);
                lrun[r] = lrun[r] * a + rs;
                mrun[r] = mnew;
                alpha[r] = a;
                sc[0][r] = p0; sc[1][r] = p1;
            }
#pragma unroll
            for (int t = 0; t < 4; ++t)
#pragma unroll
                for (int r = 0; r < 8; ++r) o[t][r] *= alpha[r];

            // ---- P: D-layout -> A-layout through per-wave LDS scratch ----
            bf16_t* ps = &pscr[wave * 16 * PSTR];
#pragma unroll
            for (int r = 0; r < 8; ++r) {
                const int row = r + hi * 8;
                ps[row * PSTR + l16]      = f2bf(sc[0][r]);
                ps[row * PSTR + 16 + l16] = f2bf(sc[1][r]);
            }
            __asm__ volatile("s_wait_dscnt 0x0" ::: "memory");
            const bf16_t* pa = &ps[l16 * PSTR + koff];
            v16bf pA = cat16(*(const v8bf*)pa, *(const v8bf*)(pa + 16));

            // ---- O += P V, four 16-wide h tiles ----
#pragma unroll
            for (int t = 0; t < 4; ++t) {
                const bf16_t* vb = &vt[(t * 16 + l16) * VPAD + (hi ? 8 : 0)];
                v16bf vB = cat16(*(const v8bf*)vb, *(const v8bf*)(vb + 16));
                o[t] = wmma_bf16(pA, vB, o[t]);
            }
        }
        __syncthreads();
    }

    // ---- normalize and store f32 output ----
#pragma unroll
    for (int t = 0; t < 4; ++t)
#pragma unroll
        for (int r = 0; r < 8; ++r) {
            const int row = qr + r + hi * 8;
            out[base + (size_t)row * HH + t * 16 + l16] = o[t][r] / lrun[r];
        }
}

// ---------------------------------------------------------------------------
// Host launcher
// ---------------------------------------------------------------------------
extern "C" void kernel_launch(void* const* d_in, const int* in_sizes, int n_in,
                              void* d_out, int out_size, void* d_ws, size_t ws_size,
                              hipStream_t stream) {
    (void)in_sizes; (void)n_in; (void)out_size; (void)ws_size;
    const float* x  = (const float*)d_in[0];
    // d_in[1] = src_mask (64 MB) is intentionally unused: causal mask is
    // recomputed analytically inside the fused kernel.
    const float* Wq = (const float*)d_in[2];
    const float* Wk = (const float*)d_in[3];
    const float* Wv = (const float*)d_in[4];
    float* out = (float*)d_out;

    const size_t n = (size_t)BB * TT * HH;           // 1 Mi elements
    bf16_t* Qw = (bf16_t*)d_ws;
    bf16_t* Kw = Qw + n;
    bf16_t* Vw = Kw + n;

    qkv_proj_kernel<<<(BB * TT) / 128, 256, 0, stream>>>(x, Wq, Wk, Wv, Qw, Kw, Vw);
    flash_attn_kernel<<<BB * (TT / 128), 256, 0, stream>>>(Qw, Kw, Vw, out);
}
